// PatchSampleF_73667279061511
// MI455X (gfx1250) — compile-verified, matrix operand order checked by reading, not run on hardware
//
#include <hip/hip_runtime.h>
#include <hip/hip_bf16.h>
#include <math.h>

typedef __attribute__((ext_vector_type(16))) _Float16 v16h;
typedef __attribute__((ext_vector_type(8)))  _Float16 v8h;
typedef __attribute__((ext_vector_type(8)))  float    v8f;

union Frag {
  v16h v;
  v8h  h[2];
};

// ---------------------------------------------------------------------------
// Prep: transpose f32 weight [C, P] (row-major, K-major) -> f16 [P, C]
// so every WMMA B-fragment lane reads contiguous 16-byte K-runs.
// ---------------------------------------------------------------------------
__global__ void wt_transpose_f16_kernel(const float* __restrict__ w,
                                        _Float16* __restrict__ wt,
                                        int C, int P) {
  int idx = blockIdx.x * blockDim.x + threadIdx.x;  // over P*C
  if (idx >= C * P) return;
  int n = idx / C;
  int c = idx - n * C;
  wt[idx] = (_Float16)w[c * P + n];
}

// ---------------------------------------------------------------------------
// ids pass-through (outputs 3 and 4 of the reference tuple)
// ---------------------------------------------------------------------------
__global__ void copy_ids_kernel(const int* __restrict__ a,
                                const int* __restrict__ b,
                                int* __restrict__ oa,
                                int* __restrict__ ob, int n) {
  int i = blockIdx.x * blockDim.x + threadIdx.x;
  if (i < n) {
    oa[i] = a[i];
    ob[i] = b[i];
  }
}

// ---------------------------------------------------------------------------
// Fused gather -> (x@W1+b1).relu -> @W2+b2 -> L2-normalize rows.
//
// One workgroup = 16 waves (512 threads) handles 16 sample rows x 256 outputs.
// Wave w computes output columns [16w, 16w+16) with v_wmma_f32_16x16x32_f16.
//
// Fragment layout (CDNA5 16-bit A 16x32 / B 32x16, wave32):
//   lane l, half = l>>4, idx = l&15
//   A: row = idx, K-runs {kb + half*8 .. +7} and {kb+16+half*8 .. +7}
//   B: col = idx, same K-runs (weights pre-transposed to [N, K] f16)
// C/D: VGPR r, lane l -> M = r + 8*half, N = idx.
// ---------------------------------------------------------------------------
template <int C>
__global__ __launch_bounds__(512) void mlp_project_kernel(
    const float* __restrict__ feat,      // [B, C, HW]
    const int* __restrict__ ids,         // [B*K] flat
    const _Float16* __restrict__ w1t,    // [256, C]  f16
    const float* __restrict__ b1,        // [256]
    const _Float16* __restrict__ w2t,    // [256, 256] f16
    const float* __restrict__ b2,        // [256]
    float* __restrict__ out,             // [B*K, 256]
    int HW, int K) {
  constexpr int P = 256;

  __shared__ __align__(16) _Float16 xsh[16 * C];  // gathered input tile (f16)
  __shared__ __align__(16) _Float16 hsh[16 * P];  // hidden tile (f16, post-ReLU)
  __shared__ float norms[16];

  const int tid     = threadIdx.x;
  const int lane    = tid & 31;
  const int wave    = tid >> 5;        // 0..15 -> which 16-column output tile
  const int half    = lane >> 4;       // 0/1: half-wave select
  const int idx16   = lane & 15;
  const int rowbase = blockIdx.x * 16; // first global sample row of this tile

  // ---- Stage 1: gather 16 rows x C channels, convert to f16 in LDS --------
  {
    const int r   = tid >> 5;          // local row 0..15 (one wave per row)
    const int row = rowbase + r;
    const int b   = row / K;
    const int id  = ids[row];          // row == b*K + k
    const float* fp = feat + (size_t)b * C * HW + id;
#pragma unroll
    for (int j = 0; j < C / 32; ++j) {
      int c = lane + 32 * j;
      xsh[r * C + c] = (_Float16)fp[(size_t)c * HW];
    }
  }
  if (tid < 16) norms[tid] = 0.0f;

  const int ncol = wave * 16 + idx16;  // this lane's output column
  // warm GL2 for second-layer weights while we crunch layer 1
  __builtin_prefetch(w2t + (size_t)ncol * P, 0, 1);

  __syncthreads();

  // ---- Stage 2: GEMM1  h = relu(x @ W1 + b1) -------------------------------
  v8f acc;
  {
    float bv = b1[ncol];
#pragma unroll
    for (int r = 0; r < 8; ++r) acc[r] = bv;
  }
#pragma unroll
  for (int kb = 0; kb < C; kb += 32) {
    Frag a, bf;
    const _Float16* ap = &xsh[idx16 * C + kb + half * 8];
    a.h[0] = *(const v8h*)(ap);
    a.h[1] = *(const v8h*)(ap + 16);
    const _Float16* bp = w1t + (size_t)ncol * C + kb + half * 8;
    bf.h[0] = *(const v8h*)(bp);
    bf.h[1] = *(const v8h*)(bp + 16);
    acc = __builtin_amdgcn_wmma_f32_16x16x32_f16(false, a.v, false, bf.v,
                                                 (short)0, acc, false, false);
  }

  // ReLU -> f16 -> LDS (row-major [16][256]) for cross-wave K exchange
#pragma unroll
  for (int r = 0; r < 8; ++r) {
    float v = acc[r] > 0.0f ? acc[r] : 0.0f;
    hsh[(r + 8 * half) * P + ncol] = (_Float16)v;
  }
  __syncthreads();

  // ---- Stage 3: GEMM2  p = h @ W2 + b2 -------------------------------------
  v8f acc2;
  {
    float bv = b2[ncol];
#pragma unroll
    for (int r = 0; r < 8; ++r) acc2[r] = bv;
  }
#pragma unroll
  for (int kb = 0; kb < P; kb += 32) {
    Frag a, bf;
    const _Float16* ap = &hsh[idx16 * P + kb + half * 8];
    a.h[0] = *(const v8h*)(ap);
    a.h[1] = *(const v8h*)(ap + 16);
    const _Float16* bp = w2t + (size_t)ncol * P + kb + half * 8;
    bf.h[0] = *(const v8h*)(bp);
    bf.h[1] = *(const v8h*)(bp + 16);
    acc2 = __builtin_amdgcn_wmma_f32_16x16x32_f16(false, a.v, false, bf.v,
                                                  (short)0, acc2, false, false);
  }

  // ---- Stage 4: row L2 norms -----------------------------------------------
  // Row M of this wave's tile lives across one 16-lane group in one VGPR:
  // reduce squares across the group, then one ds atomic per (row, wave).
#pragma unroll
  for (int r = 0; r < 8; ++r) {
    float s = acc2[r] * acc2[r];
    s += __shfl_xor(s, 1, 32);
    s += __shfl_xor(s, 2, 32);
    s += __shfl_xor(s, 4, 32);
    s += __shfl_xor(s, 8, 32);
    if (idx16 == 0) atomicAdd(&norms[r + 8 * half], s);
  }
  __syncthreads();
  if (tid < 16) norms[tid] = 1.0f / fmaxf(sqrtf(norms[tid]), 1e-12f);
  __syncthreads();

  // ---- Stage 5: scale + store ---------------------------------------------
#pragma unroll
  for (int r = 0; r < 8; ++r) {
    int m = r + 8 * half;
    out[(size_t)(rowbase + m) * P + ncol] = acc2[r] * norms[m];
  }
}

// ---------------------------------------------------------------------------
extern "C" void kernel_launch(void* const* d_in, const int* in_sizes, int n_in,
                              void* d_out, int out_size, void* d_ws,
                              size_t ws_size, hipStream_t stream) {
  const float* feat0 = (const float*)d_in[0];   // [8,256,128,128]
  const float* feat1 = (const float*)d_in[1];   // [8,512,64,64]
  const int*   ids0  = (const int*)d_in[2];     // [8,4096]
  const int*   ids1  = (const int*)d_in[3];     // [8,4096]
  const float* w1_0  = (const float*)d_in[4];   // [256,256]
  const float* b1_0  = (const float*)d_in[5];
  const float* w2_0  = (const float*)d_in[6];   // [256,256]
  const float* b2_0  = (const float*)d_in[7];
  const float* w1_1  = (const float*)d_in[8];   // [512,256]
  const float* b1_1  = (const float*)d_in[9];
  const float* w2_1  = (const float*)d_in[10];  // [256,256]
  const float* b2_1  = (const float*)d_in[11];

  // workspace: transposed f16 weights (~640 KB)
  _Float16* ws   = (_Float16*)d_ws;
  _Float16* w1t0 = ws;                                // [256,256]
  _Float16* w2t0 = w1t0 + 256 * 256;                  // [256,256]
  _Float16* w1t1 = w2t0 + 256 * 256;                  // [256,512]
  _Float16* w2t1 = w1t1 + 256 * 512;                  // [256,256]

  {
    int n = 256 * 256;
    wt_transpose_f16_kernel<<<(n + 255) / 256, 256, 0, stream>>>(w1_0, w1t0,
                                                                 256, 256);
    wt_transpose_f16_kernel<<<(n + 255) / 256, 256, 0, stream>>>(w2_0, w2t0,
                                                                 256, 256);
    int n1 = 512 * 256;
    wt_transpose_f16_kernel<<<(n1 + 255) / 256, 256, 0, stream>>>(w1_1, w1t1,
                                                                  512, 256);
    wt_transpose_f16_kernel<<<(n + 255) / 256, 256, 0, stream>>>(w2_1, w2t1,
                                                                 256, 256);
  }

  const int BK = 8 * 4096;             // 32768 sample rows per stream
  float* out0 = (float*)d_out;                         // proj0 [32768,256]
  float* out1 = out0 + (size_t)BK * 256;               // proj1 [32768,256]
  int*   oid  = (int*)(out1 + (size_t)BK * 256);       // ids0, ids1

  mlp_project_kernel<256><<<BK / 16, 512, 0, stream>>>(
      feat0, ids0, w1t0, b1_0, w2t0, b2_0, out0, 128 * 128, 4096);
  mlp_project_kernel<512><<<BK / 16, 512, 0, stream>>>(
      feat1, ids1, w1t1, b1_1, w2t1, b2_1, out1, 64 * 64, 4096);

  copy_ids_kernel<<<(BK + 255) / 256, 256, 0, stream>>>(ids0, ids1, oid,
                                                        oid + BK, BK);
}
